// GRU_57896159150705
// MI455X (gfx1250) — matrix operand order repaired and checked by reading, not compiled, not run on hardware
//
#include <hip/hip_runtime.h>
#include <hip/hip_bf16.h>

// ---------------------------------------------------------------------------
// 2-layer GRU for MI455X (gfx1250): persistent kernel, bf16 WMMA, fp32 accum,
// weights LDS-resident per WGP (168 KB slice of the 320 KB LDS), bf16 shadow
// hidden state so the recurrence inner loop has zero fp32->bf16 conversions.
// B=64, S=1024, I=256, H=512, L=2
// ---------------------------------------------------------------------------

#define GB 64
#define GS 1024
#define GI 256
#define GH 512

typedef __attribute__((ext_vector_type(16))) __bf16 v16bf;
typedef __attribute__((ext_vector_type(8)))  __bf16 v8bf;
typedef __attribute__((ext_vector_type(8)))  float  v8f;

// ---------------- packed-weight geometry ------------------------------------
// B-operand pack: [gate(3)][ntile(32)][ktile(K/32)][lane(32)][16 bf16]
// lane l: column n = ntile*16 + (l&15); k = ktile*32 + (l>>4)*16 + e
#define W0_KT   (GI / 32)   // 8
#define U_KT    (GH / 32)   // 16
#define W0_ELEMS (3 * 32 * W0_KT * 32 * 16)  // 393216
#define U_ELEMS  (3 * 32 * U_KT  * 32 * 16)  // 786432

// ---------------- LDS layout (bf16 elements) --------------------------------
// Per-block (= per nt) slice of all four weight groups, gate-major.
#define LDS_W0   0
#define LDS_U0   (LDS_W0 + 3 * W0_KT * 512)   // 12288
#define LDS_W1   (LDS_U0 + 3 * U_KT * 512)    // 36864
#define LDS_U1   (LDS_W1 + 3 * U_KT * 512)    // 61440
#define LDS_ELEMS (LDS_U1 + 3 * U_KT * 512)   // 86016 elems = 172032 bytes
#define LDS_BYTES (LDS_ELEMS * 2)

// ---------------- workspace layout (bytes) ----------------------------------
#define WS_BAR_OFF   0                                    // barrier, 256 B
#define WS_H0_OFF    256                                  // fp32 h0, 2 parities
#define WS_H1_OFF    (WS_H0_OFF  + 2 * GB * GH * 4)       // fp32 h1
#define WS_HB0_OFF   (WS_H1_OFF  + 2 * GB * GH * 4)       // bf16 h0 shadow
#define WS_HB1_OFF   (WS_HB0_OFF + 2 * GB * GH * 2)       // bf16 h1 shadow
#define WS_W0_OFF    (WS_HB1_OFF + 2 * GB * GH * 2)
#define WS_U0_OFF    (WS_W0_OFF  + W0_ELEMS * 2)
#define WS_W1_OFF    (WS_U0_OFF  + U_ELEMS * 2)
#define WS_U1_OFF    (WS_W1_OFF  + U_ELEMS * 2)
#define WS_XB_OFF    (WS_U1_OFF  + U_ELEMS * 2)           // bf16 x mirror, 32 MB

// ---------------------------------------------------------------------------
__device__ __forceinline__ v8f wmma_bf16(v16bf a, v16bf b, v8f c) {
  return __builtin_amdgcn_wmma_f32_16x16x32_bf16(false, a, false, b,
                                                 (short)0, c, false, false);
}

__device__ __forceinline__ float fast_sigmoid(float x) {
  return 1.0f / (1.0f + __expf(-x));
}
__device__ __forceinline__ float fast_tanh(float x) {
  return 2.0f / (1.0f + __expf(-2.0f * x)) - 1.0f;
}

// 16x32 bf16 A-fragment from a bf16 row-major row.
// Lane l (rows: l&15): K = {k0..k0+15}, k0 = kt*32 + 16*(l>>4).
__device__ __forceinline__ v16bf load_a_row_bf16(const __bf16* rowbase, int k0) {
  v8bf lo = *(const v8bf*)(rowbase + k0);        // 16B global load
  v8bf hi = *(const v8bf*)(rowbase + k0 + 8);    // 16B global load
  v16bf a;
#pragma unroll
  for (int i = 0; i < 8; ++i) { a[i] = lo[i]; a[i + 8] = hi[i]; }
  return a;
}

// B-fragment (32x16 bf16) from the LDS-resident weight slice.
__device__ __forceinline__ v16bf load_b_lds(const __bf16* sect, int kt, int lane) {
  return *(const v16bf*)(sect + ((size_t)kt * 32 + lane) * 16);  // 32B ds load
}

// Cooperative copy of a contiguous bf16 run into LDS (uint4 granularity).
__device__ __forceinline__ void stage_slice(const __bf16* gsrc, __bf16* ldst,
                                            int nelems) {
  const uint4* s = (const uint4*)gsrc;
  uint4* d = (uint4*)ldst;
  int nvec = nelems >> 3;   // 8 bf16 per uint4
  for (int i = threadIdx.x; i < nvec; i += blockDim.x) d[i] = s[i];
}

// Device-wide barrier (persistent grid, agent-scope atomics, s_sleep spin).
__device__ __forceinline__ void grid_barrier(unsigned* bar, unsigned nblocks,
                                             unsigned& gen) {
  __syncthreads();
  if (threadIdx.x == 0) {
    __threadfence();
    unsigned g = gen;
    unsigned arrived = __hip_atomic_fetch_add(&bar[0], 1u, __ATOMIC_ACQ_REL,
                                              __HIP_MEMORY_SCOPE_AGENT);
    if (arrived == nblocks - 1) {
      __hip_atomic_store(&bar[0], 0u, __ATOMIC_RELAXED, __HIP_MEMORY_SCOPE_AGENT);
      __hip_atomic_store(&bar[1], g + 1, __ATOMIC_RELEASE, __HIP_MEMORY_SCOPE_AGENT);
    } else {
      while (__hip_atomic_load(&bar[1], __ATOMIC_ACQUIRE,
                               __HIP_MEMORY_SCOPE_AGENT) <= g) {
        __builtin_amdgcn_s_sleep(2);
      }
    }
    __threadfence();
  }
  gen++;
  __syncthreads();
}

// ---------------------------------------------------------------------------
// Pack fp32 [K,512] weights (3 gates: r,z,h) into the bf16 B-operand layout.
__global__ void gru_pack_weights(const float* __restrict__ sr,
                                 const float* __restrict__ sz,
                                 const float* __restrict__ sh,
                                 __bf16* __restrict__ dst, int Kt) {
  int total = 3 * 32 * Kt * 32 * 16;
  for (int tid = blockIdx.x * blockDim.x + threadIdx.x; tid < total;
       tid += gridDim.x * blockDim.x) {
    int e    = tid & 15;
    int lane = (tid >> 4) & 31;
    int kt   = (tid >> 9) % Kt;
    int rest = (tid >> 9) / Kt;
    int nt   = rest & 31;
    int gate = rest >> 5;
    int n = nt * 16 + (lane & 15);
    int k = kt * 32 + (lane >> 4) * 16 + e;
    const float* src = (gate == 0) ? sr : ((gate == 1) ? sz : sh);
    dst[tid] = (__bf16)src[(size_t)k * GH + n];
  }
}

// Convert x [B,S,I] fp32 -> bf16 mirror (same layout). One bandwidth pass.
__global__ void gru_cvt_x(const float* __restrict__ x, __bf16* __restrict__ xb,
                          int n) {
  for (int i = (blockIdx.x * blockDim.x + threadIdx.x) * 8; i < n;
       i += gridDim.x * blockDim.x * 8) {
    float4 f0 = ((const float4*)(x + i))[0];
    float4 f1 = ((const float4*)(x + i))[1];
    v8bf o;
    o[0] = (__bf16)f0.x; o[1] = (__bf16)f0.y; o[2] = (__bf16)f0.z; o[3] = (__bf16)f0.w;
    o[4] = (__bf16)f1.x; o[5] = (__bf16)f1.y; o[6] = (__bf16)f1.z; o[7] = (__bf16)f1.w;
    *(v8bf*)(xb + i) = o;
  }
}

// ---------------------------------------------------------------------------
// Persistent GRU scan. 32 blocks x 128 threads; block = nt (16 columns),
// wave = mt (16 rows). Weight slice (168 KB) staged into LDS once; recurrence
// reads B-operands from LDS only.
__global__ void __launch_bounds__(128)
gru_persistent(const __bf16* __restrict__ xb,      // [B,S,I] bf16 mirror
               const float* __restrict__ h_init,   // [2,B,H]
               const float* __restrict__ br0, const float* __restrict__ bz0,
               const float* __restrict__ bh0,
               const float* __restrict__ br1, const float* __restrict__ bz1,
               const float* __restrict__ bh1,
               const __bf16* __restrict__ W0p, const __bf16* __restrict__ U0p,
               const __bf16* __restrict__ W1p, const __bf16* __restrict__ U1p,
               float*  __restrict__ h0buf,         // fp32 h0, 2 parities
               float*  __restrict__ h1buf,         // fp32 h1, 2 parities
               __bf16* __restrict__ hb0buf,        // bf16 h0 shadow, 2 parities
               __bf16* __restrict__ hb1buf,        // bf16 h1 shadow, 2 parities
               unsigned* __restrict__ bar,
               float* __restrict__ out)            // [B,S,H] ++ [2,B,H]
{
  extern __shared__ __align__(16) char smem_raw[];
  __bf16* lds = (__bf16*)smem_raw;

  const int lane  = threadIdx.x & 31;
  const int mt    = threadIdx.x >> 5;          // wave id = mt (0..3)
  const int nt    = blockIdx.x;                // block id = nt (0..31)
  const int mbase = mt * 16;
  const int khalf = lane >> 4;
  const int row0  = khalf * 8;
  const int col   = nt * 16 + (lane & 15);
  const int mrow0 = mbase + row0;
  const unsigned nblocks = gridDim.x;
  unsigned gen = 0;

  // ---- stage this block's weight slice into LDS (once; time-invariant) ----
#pragma unroll
  for (int g = 0; g < 3; ++g) {
    stage_slice(W0p + (size_t)((g * 32 + nt) * W0_KT) * 512,
                lds + LDS_W0 + g * W0_KT * 512, W0_KT * 512);
    stage_slice(U0p + (size_t)((g * 32 + nt) * U_KT) * 512,
                lds + LDS_U0 + g * U_KT * 512, U_KT * 512);
    stage_slice(W1p + (size_t)((g * 32 + nt) * U_KT) * 512,
                lds + LDS_W1 + g * U_KT * 512, U_KT * 512);
    stage_slice(U1p + (size_t)((g * 32 + nt) * U_KT) * 512,
                lds + LDS_U1 + g * U_KT * 512, U_KT * 512);
  }

  // ---- init h state (fp32 + bf16 shadow), parity 0 ----
#pragma unroll
  for (int i = 0; i < 8; ++i) {
    int m = mrow0 + i;
    float v0 = h_init[(size_t)0 * GB * GH + m * GH + col];
    float v1 = h_init[(size_t)1 * GB * GH + m * GH + col];
    h0buf[m * GH + col]  = v0;  hb0buf[m * GH + col] = (__bf16)v0;
    h1buf[m * GH + col]  = v1;  hb1buf[m * GH + col] = (__bf16)v1;
  }
  const float br0v = br0[col], bz0v = bz0[col], bh0v = bh0[col];
  const float br1v = br1[col], bz1v = bz1[col], bh1v = bh1[col];
  __syncthreads();                      // LDS staging visible in-block
  grid_barrier(bar, nblocks, gen);      // h-state visible device-wide

  const __bf16* xrow = xb + (size_t)(mbase + (lane & 15)) * GS * GI;
  const __bf16* ldsW0 = lds + LDS_W0;
  const __bf16* ldsU0 = lds + LDS_U0;
  const __bf16* ldsW1 = lds + LDS_W1;
  const __bf16* ldsU1 = lds + LDS_U1;

  for (int t = 0; t < GS; ++t) {
    const size_t p0 = (size_t)(t & 1) * GB * GH;
    const size_t p1 = (size_t)((t + 1) & 1) * GB * GH;
    const float*  h0_prev  = h0buf  + p0;  float*  h0_next  = h0buf  + p1;
    const float*  h1_prev  = h1buf  + p0;  float*  h1_next  = h1buf  + p1;
    const __bf16* hb0_prev = hb0buf + p0;  __bf16* hb0_next = hb0buf + p1;
    const __bf16* hb1_prev = hb1buf + p0;  __bf16* hb1_next = hb1buf + p1;
    const __bf16* xt = xrow + (size_t)t * GI;
    if (t + 1 < GS) __builtin_prefetch(xt + GI, 0, 0);   // next step's x row

    // ===================== layer 0 =====================
    v8f ar = {}, az = {}, axh = {}, ahh = {};
    for (int kt = 0; kt < W0_KT; ++kt) {               // x-part, K=256
      v16bf a  = load_a_row_bf16(xt, kt * 32 + 16 * khalf);
      v16bf b0 = load_b_lds(ldsW0 + 0 * W0_KT * 512, kt, lane);
      v16bf b1 = load_b_lds(ldsW0 + 1 * W0_KT * 512, kt, lane);
      v16bf b2 = load_b_lds(ldsW0 + 2 * W0_KT * 512, kt, lane);
      ar  = wmma_bf16(a, b0, ar);
      az  = wmma_bf16(a, b1, az);
      axh = wmma_bf16(a, b2, axh);
    }
    {
      const __bf16* hrow = hb0_prev + (size_t)(mbase + (lane & 15)) * GH;
      for (int kt = 0; kt < U_KT; ++kt) {              // h-part, K=512
        v16bf a  = load_a_row_bf16(hrow, kt * 32 + 16 * khalf);
        v16bf b0 = load_b_lds(ldsU0 + 0 * U_KT * 512, kt, lane);
        v16bf b1 = load_b_lds(ldsU0 + 1 * U_KT * 512, kt, lane);
        v16bf b2 = load_b_lds(ldsU0 + 2 * U_KT * 512, kt, lane);
        ar  = wmma_bf16(a, b0, ar);
        az  = wmma_bf16(a, b1, az);
        ahh = wmma_bf16(a, b2, ahh);
      }
    }
#pragma unroll
    for (int i = 0; i < 8; ++i) {
      int m = mrow0 + i;
      float hp = h0_prev[m * GH + col];
      float r  = fast_sigmoid(ar[i] + br0v);
      float z  = fast_sigmoid(az[i] + bz0v);
      float hh = fast_tanh(axh[i] + bh0v + r * ahh[i]);
      float hn = (1.0f - z) * hp + z * hh;
      h0_next[m * GH + col]  = hn;
      hb0_next[m * GH + col] = (__bf16)hn;
    }
    grid_barrier(bar, nblocks, gen);

    // ===================== layer 1 =====================
    v8f cr = {}, cz = {}, cxh = {}, chh = {};
    {
      const __bf16* arow = hb0_next + (size_t)(mbase + (lane & 15)) * GH;
      for (int kt = 0; kt < U_KT; ++kt) {              // W1-part (input = hn0)
        v16bf a  = load_a_row_bf16(arow, kt * 32 + 16 * khalf);
        v16bf b0 = load_b_lds(ldsW1 + 0 * U_KT * 512, kt, lane);
        v16bf b1 = load_b_lds(ldsW1 + 1 * U_KT * 512, kt, lane);
        v16bf b2 = load_b_lds(ldsW1 + 2 * U_KT * 512, kt, lane);
        cr  = wmma_bf16(a, b0, cr);
        cz  = wmma_bf16(a, b1, cz);
        cxh = wmma_bf16(a, b2, cxh);
      }
      const __bf16* hrow = hb1_prev + (size_t)(mbase + (lane & 15)) * GH;
      for (int kt = 0; kt < U_KT; ++kt) {              // U1-part (recurrent)
        v16bf a  = load_a_row_bf16(hrow, kt * 32 + 16 * khalf);
        v16bf b0 = load_b_lds(ldsU1 + 0 * U_KT * 512, kt, lane);
        v16bf b1 = load_b_lds(ldsU1 + 1 * U_KT * 512, kt, lane);
        v16bf b2 = load_b_lds(ldsU1 + 2 * U_KT * 512, kt, lane);
        cr  = wmma_bf16(a, b0, cr);
        cz  = wmma_bf16(a, b1, cz);
        chh = wmma_bf16(a, b2, chh);
      }
    }
#pragma unroll
    for (int i = 0; i < 8; ++i) {
      int m = mrow0 + i;
      float hp = h1_prev[m * GH + col];
      float r  = fast_sigmoid(cr[i] + br1v);
      float z  = fast_sigmoid(cz[i] + bz1v);
      float hh = fast_tanh(cxh[i] + bh1v + r * chh[i]);
      float hn = (1.0f - z) * hp + z * hh;
      h1_next[m * GH + col]  = hn;
      hb1_next[m * GH + col] = (__bf16)hn;
      out[((size_t)m * GS + t) * GH + col] = hn;       // outputs[B,S,H]
    }
    grid_barrier(bar, nblocks, gen);
  }

  // final states: last write landed in parity-0 buffers (S even)
  float* outh = out + (size_t)GB * GS * GH;
#pragma unroll
  for (int i = 0; i < 8; ++i) {
    int m = mrow0 + i;
    outh[m * GH + col]           = h0buf[m * GH + col];
    outh[GB * GH + m * GH + col] = h1buf[m * GH + col];
  }
}

// ---------------------------------------------------------------------------
extern "C" void kernel_launch(void* const* d_in, const int* in_sizes, int n_in,
                              void* d_out, int out_size, void* d_ws,
                              size_t ws_size, hipStream_t stream) {
  (void)in_sizes; (void)n_in; (void)out_size; (void)ws_size;
  // 0:x 1:h0 2:Wz0 3:Uz0 4:bz0 5:Wr0 6:Ur0 7:br0 8:Wh0 9:Uh0 10:bh0
  // 11:Wz1 12:Uz1 13:bz1 14:Wr1 15:Ur1 16:br1 17:Wh1 18:Uh1 19:bh1
  const float* x   = (const float*)d_in[0];
  const float* h0  = (const float*)d_in[1];
  const float* Wz0 = (const float*)d_in[2];
  const float* Uz0 = (const float*)d_in[3];
  const float* bz0 = (const float*)d_in[4];
  const float* Wr0 = (const float*)d_in[5];
  const float* Ur0 = (const float*)d_in[6];
  const float* br0 = (const float*)d_in[7];
  const float* Wh0 = (const float*)d_in[8];
  const float* Uh0 = (const float*)d_in[9];
  const float* bh0 = (const float*)d_in[10];
  const float* Wz1 = (const float*)d_in[11];
  const float* Uz1 = (const float*)d_in[12];
  const float* bz1 = (const float*)d_in[13];
  const float* Wr1 = (const float*)d_in[14];
  const float* Ur1 = (const float*)d_in[15];
  const float* br1 = (const float*)d_in[16];
  const float* Wh1 = (const float*)d_in[17];
  const float* Uh1 = (const float*)d_in[18];
  const float* bh1 = (const float*)d_in[19];

  char* ws = (char*)d_ws;
  unsigned* bar   = (unsigned*)(ws + WS_BAR_OFF);
  float*   h0buf  = (float*)(ws + WS_H0_OFF);
  float*   h1buf  = (float*)(ws + WS_H1_OFF);
  __bf16*  hb0buf = (__bf16*)(ws + WS_HB0_OFF);
  __bf16*  hb1buf = (__bf16*)(ws + WS_HB1_OFF);
  __bf16*  W0p    = (__bf16*)(ws + WS_W0_OFF);
  __bf16*  U0p    = (__bf16*)(ws + WS_U0_OFF);
  __bf16*  W1p    = (__bf16*)(ws + WS_W1_OFF);
  __bf16*  U1p    = (__bf16*)(ws + WS_U1_OFF);
  __bf16*  xbuf   = (__bf16*)(ws + WS_XB_OFF);

  hipMemsetAsync(bar, 0, 256, stream);  // barrier state must start at zero

  // One-time prep: bf16 x mirror + packed bf16 weights (gate order r,z,h)
  gru_cvt_x<<<4096, 256, 0, stream>>>(x, xbuf, GB * GS * GI);
  gru_pack_weights<<<512, 256, 0, stream>>>(Wr0, Wz0, Wh0, W0p, W0_KT);
  gru_pack_weights<<<512, 256, 0, stream>>>(Ur0, Uz0, Uh0, U0p, U_KT);
  gru_pack_weights<<<512, 256, 0, stream>>>(Wr1, Wz1, Wh1, W1p, U_KT);
  gru_pack_weights<<<512, 256, 0, stream>>>(Ur1, Uz1, Uh1, U1p, U_KT);

  // Persistent scan: 32 blocks x 128 threads, 168 KB dynamic LDS per block
  // (weights LDS-resident on each WGP; 320 KB available per WGP on CDNA5).
  gru_persistent<<<32, 128, LDS_BYTES, stream>>>(
      xbuf, h0, br0, bz0, bh0, br1, bz1, bh1,
      W0p, U0p, W1p, U1p, h0buf, h1buf, hb0buf, hb1buf, bar, (float*)d_out);
}
